// BlockAttention_63256278335987
// MI455X (gfx1250) — matrix operand order repaired and checked by reading, not compiled, not run on hardware
//
#include <hip/hip_runtime.h>

typedef _Float16 v16h __attribute__((ext_vector_type(16)));
typedef _Float16 h8   __attribute__((ext_vector_type(8)));
typedef float    v8f  __attribute__((ext_vector_type(8)));
typedef float    f4   __attribute__((ext_vector_type(4)));

#define LDX 200   // f16 row stride for 64x192 token-major buffers (400B, 16B aligned)
#define LDV 72    // f16 row stride for Vt [192][64] channel-major (144B, 16B aligned)
#define LDA 72    // f16 row stride for per-pair prob buffer [64][64]

__device__ __forceinline__ v8f wmma_f16(v16h a, v16h b, v8f c) {
  // D = A(16x32 f16) x B(32x16 f16) + C(16x16 f32)
  return __builtin_amdgcn_wmma_f32_16x16x32_f16(false, a, false, b, (short)0, c,
                                                false, false);
}

// A-fragment (16x32, f16) from row-major LDS matrix: lane = M row, K pairs per
// contiguous layout: element s -> K = half*16 + s.
__device__ __forceinline__ v16h load_a_frag(const _Float16* base, int ld,
                                            int mBase, int kBase, int lane) {
  const int m  = lane & 15;
  const int hf = lane >> 4;
  const _Float16* row = base + (mBase + m) * ld + kBase + hf * 16;
  h8 lo = *(const h8*)(row);        // K = hf*16 .. hf*16+7
  h8 hi = *(const h8*)(row + 8);    // K = hf*16+8 .. hf*16+15
  v16h a;
#pragma unroll
  for (int i = 0; i < 8; ++i) { a[i] = lo[i]; a[8 + i] = hi[i]; }
  return a;
}

// B-fragment (32x16, f16) where B[k][n] = M[n][k] for row-major source M:
// lane = N, VGPR v -> K = half*16 + 2v  ==> one contiguous 16-element row read.
__device__ __forceinline__ v16h load_b_frag_lds(const _Float16* base, int ld,
                                                int nBase, int kBase, int lane) {
  const int n  = lane & 15;
  const int hf = lane >> 4;
  const _Float16* row = base + (nBase + n) * ld + kBase + hf * 16;
  h8 lo = *(const h8*)(row);
  h8 hi = *(const h8*)(row + 8);
  v16h b;
#pragma unroll
  for (int i = 0; i < 8; ++i) { b[i] = lo[i]; b[8 + i] = hi[i]; }
  return b;
}

// Same, but source is a row-major f32 weight matrix in global memory (L2-hot).
__device__ __forceinline__ v16h load_b_frag_w(const float* __restrict__ W, int ldw,
                                              int nBase, int kBase, int lane) {
  const int n  = lane & 15;
  const int hf = lane >> 4;
  const float* row = W + (nBase + n) * ldw + kBase + hf * 16;
  const f4* r4 = (const f4*)row;
  f4 c0 = r4[0], c1 = r4[1], c2 = r4[2], c3 = r4[3];
  v16h b;
#pragma unroll
  for (int i = 0; i < 4; ++i) {
    b[i]      = (_Float16)c0[i];
    b[4 + i]  = (_Float16)c1[i];
    b[8 + i]  = (_Float16)c2[i];
    b[12 + i] = (_Float16)c3[i];
  }
  return b;
}

__global__ __launch_bounds__(256) void block_attn_kernel(
    const float* __restrict__ x,
    const float* __restrict__ wqkv,
    const float* __restrict__ bqkv,
    const float* __restrict__ wout,
    const float* __restrict__ bout,
    float* __restrict__ out) {
  constexpr int C = 192, HW = 256, BS = 8, S = 64, HD = 32;

  __shared__ __align__(16) _Float16 sX[S * LDX];        // window (f16); reused as attn output
  __shared__ __align__(16) _Float16 sQ[S * LDX];        // Q (pre-scaled), token-major
  __shared__ __align__(16) _Float16 sK[S * LDX];        // K, token-major
  __shared__ __align__(16) _Float16 sVt[C * LDV];       // V transposed: [channel][token]
  __shared__ __align__(16) _Float16 sAttn[3 * S * LDA]; // per wave-pair probs (f16)

  const int blk  = blockIdx.x;
  const int b    = blk >> 10;
  const int hb   = (blk >> 5) & 31;
  const int wb   = blk & 31;
  const int tid  = threadIdx.x;
  const int wave = tid >> 5;
  const int lane = tid & 31;

  // offset of element (c=0, i=0, j=0) of this window in x / out ([B,C,H,W])
  const int base0 = ((b * C) * HW + hb * BS) * HW + wb * BS;

  // ---------------- Phase 0: gather window -> LDS f16; prefetch weights ------
  for (int p = tid; p < (3 * C * C) / 32; p += 256)
    __builtin_prefetch(wqkv + p * 32, 0, 2);            // global_prefetch_b8
  for (int idx = tid; idx < S * C; idx += 256) {
    const int c = idx >> 6;
    const int t = idx & 63;
    const float v = x[base0 + c * (HW * HW) + (t >> 3) * HW + (t & 7)];
    sX[t * LDX + c] = (_Float16)v;
  }
  __syncthreads();

  // ---------------- Phase 1: QKV = X @ Wqkv^T + b  (576 outputs = 36 N-tiles)
  const float qscale = 0.17677669529663689f;  // 32^-0.5
  for (int nt = wave; nt < 36; nt += 8) {
    v8f acc[4] = {};
#pragma unroll
    for (int kt = 0; kt < 6; ++kt) {
      v16h bf = load_b_frag_w(wqkv, C, nt * 16, kt * 32, lane);
#pragma unroll
      for (int mt = 0; mt < 4; ++mt) {
        v16h af = load_a_frag(sX, LDX, mt * 16, kt * 32, lane);
        acc[mt] = wmma_f16(af, bf, acc[mt]);
      }
    }
    const int n  = lane & 15;
    const int hf = lane >> 4;
    const int o  = nt * 16 + n;
    const float bias = bqkv[o];
    if (nt < 12) {                      // Q region, pre-scale
#pragma unroll
      for (int mt = 0; mt < 4; ++mt)
#pragma unroll
        for (int r = 0; r < 8; ++r)
          sQ[(mt * 16 + hf * 8 + r) * LDX + o] =
              (_Float16)((acc[mt][r] + bias) * qscale);
    } else if (nt < 24) {               // K region
      const int c = o - C;
#pragma unroll
      for (int mt = 0; mt < 4; ++mt)
#pragma unroll
        for (int r = 0; r < 8; ++r)
          sK[(mt * 16 + hf * 8 + r) * LDX + c] = (_Float16)(acc[mt][r] + bias);
    } else {                            // V region -> store transposed, vectorized
      const int c = o - 2 * C;
#pragma unroll
      for (int mt = 0; mt < 4; ++mt) {
        h8 vv;
#pragma unroll
        for (int r = 0; r < 8; ++r) vv[r] = (_Float16)(acc[mt][r] + bias);
        *(h8*)(&sVt[c * LDV + mt * 16 + hf * 8]) = vv;  // tokens contiguous
      }
    }
  }
  __syncthreads();

  // ---------------- Phase 2: attention (waves 0..5; wave pair = head pair) ---
  if (wave < 6) {
    const int pair  = wave >> 1;   // prob buffer id; heads {pair, pair+3}
    const int mhalf = wave & 1;    // which 32 query rows this wave owns
    _Float16* sA = sAttn + pair * (S * LDA);
    const int n  = lane & 15;
    const int hf = lane >> 4;
#pragma unroll
    for (int pass = 0; pass < 2; ++pass) {
      const int h = pass * 3 + pair;
      // ---- scores = Q_h @ K_h^T  (K dim = hd = 32 -> single WMMA per tile)
      v16h aq0 = load_a_frag(sQ, LDX, (mhalf * 2 + 0) * 16, h * HD, lane);
      v16h aq1 = load_a_frag(sQ, LDX, (mhalf * 2 + 1) * 16, h * HD, lane);
      v8f s[2][4];
#pragma unroll
      for (int nt = 0; nt < 4; ++nt) {
        v16h bk = load_b_frag_lds(sK, LDX, nt * 16, h * HD, lane);
        v8f z = {};
        s[0][nt] = wmma_f16(aq0, bk, z);
        s[1][nt] = wmma_f16(aq1, bk, z);
      }
      // ---- row softmax: row m = mt*16 + hf*8 + r lives in this 16-lane half
      float rinv[2][8];
#pragma unroll
      for (int mi = 0; mi < 2; ++mi) {
#pragma unroll
        for (int r = 0; r < 8; ++r) {
          float mx = fmaxf(fmaxf(s[mi][0][r], s[mi][1][r]),
                           fmaxf(s[mi][2][r], s[mi][3][r]));
          mx = fmaxf(mx, __shfl_xor(mx, 1, 32));
          mx = fmaxf(mx, __shfl_xor(mx, 2, 32));
          mx = fmaxf(mx, __shfl_xor(mx, 4, 32));
          mx = fmaxf(mx, __shfl_xor(mx, 8, 32));
          float sum = 0.f;
#pragma unroll
          for (int nt = 0; nt < 4; ++nt) {
            float e = __expf(s[mi][nt][r] - mx);
            s[mi][nt][r] = e;          // unnormalized prob
            sum += e;
          }
          sum += __shfl_xor(sum, 1, 32);
          sum += __shfl_xor(sum, 2, 32);
          sum += __shfl_xor(sum, 4, 32);
          sum += __shfl_xor(sum, 8, 32);
          rinv[mi][r] = 1.0f / sum;
        }
      }
      // ---- stash unnormalized probs as f16 in A-source (row-major) layout
#pragma unroll
      for (int mi = 0; mi < 2; ++mi) {
        const int mt = mhalf * 2 + mi;
#pragma unroll
        for (int nt = 0; nt < 4; ++nt)
#pragma unroll
          for (int r = 0; r < 8; ++r)
            sA[(mt * 16 + hf * 8 + r) * LDA + nt * 16 + n] =
                (_Float16)s[mi][nt][r];
      }
      // ---- P @ V_h, normalize with register-resident 1/rowsum
#pragma unroll
      for (int mi = 0; mi < 2; ++mi) {
        const int mt = mhalf * 2 + mi;
#pragma unroll
        for (int nt2 = 0; nt2 < 2; ++nt2) {
          v8f acc = {};
#pragma unroll
          for (int kt = 0; kt < 2; ++kt) {
            v16h ap = load_a_frag(sA, LDA, mt * 16, kt * 32, lane);
            v16h bv = load_b_frag_lds(sVt, LDV, h * HD + nt2 * 16, kt * 32, lane);
            acc = wmma_f16(ap, bv, acc);
          }
          const int cch = h * HD + nt2 * 16 + n;
#pragma unroll
          for (int r = 0; r < 8; ++r)
            sX[(mt * 16 + hf * 8 + r) * LDX + cch] =
                (_Float16)(acc[r] * rinv[mi][r]);   // sX reused as attn output
        }
      }
    }
  }
  __syncthreads();

  // ---------------- Phase 3: out = AO @ Wout^T + b, scatter with inv-permute -
  for (int nt = wave; nt < 12; nt += 8) {
    v8f acc[4] = {};
#pragma unroll
    for (int kt = 0; kt < 6; ++kt) {
      v16h bf = load_b_frag_w(wout, C, nt * 16, kt * 32, lane);
#pragma unroll
      for (int mt = 0; mt < 4; ++mt) {
        v16h af = load_a_frag(sX, LDX, mt * 16, kt * 32, lane);
        acc[mt] = wmma_f16(af, bf, acc[mt]);
      }
    }
    const int n  = lane & 15;
    const int hf = lane >> 4;
    const int c  = nt * 16 + n;
    const float bias = bout[c];
    float* op = out + base0 + c * (HW * HW);
#pragma unroll
    for (int mt = 0; mt < 4; ++mt)
#pragma unroll
      for (int r = 0; r < 8; ++r) {
        const int t = mt * 16 + hf * 8 + r;
        op[(t >> 3) * HW + (t & 7)] = acc[mt][r] + bias;
      }
  }
}

extern "C" void kernel_launch(void* const* d_in, const int* in_sizes, int n_in,
                              void* d_out, int out_size, void* d_ws, size_t ws_size,
                              hipStream_t stream) {
  (void)in_sizes; (void)n_in; (void)d_ws; (void)ws_size; (void)out_size;
  const float* x    = (const float*)d_in[0];
  const float* wqkv = (const float*)d_in[1];
  const float* bqkv = (const float*)d_in[2];
  const float* wout = (const float*)d_in[3];
  const float* bout = (const float*)d_in[4];
  float* out = (float*)d_out;
  hipLaunchKernelGGL(block_attn_kernel, dim3(4096), dim3(256), 0, stream,
                     x, wqkv, bqkv, wout, bout, out);
}
